// ComEnc_17197049053642
// MI455X (gfx1250) — matrix operand order repaired and checked by reading, not compiled, not run on hardware
//
#include <hip/hip_runtime.h>
#include <math.h>

// ---------------- WMMA types ----------------
typedef __attribute__((ext_vector_type(16))) __bf16 v16bf;
typedef __attribute__((ext_vector_type(8)))  float  v8f;

union BF16x16 {
    v16bf v;
    unsigned short s[16];
    uint4 q[2];
};

__device__ __forceinline__ unsigned short f2bf(float f) {
    unsigned int u = __float_as_uint(f);
    unsigned int r = (u + 0x7FFFu + ((u >> 16) & 1u)) >> 16;   // round-to-nearest-even
    return (unsigned short)r;
}

// ---------------- constants ----------------
#define N_NODES 131072
#define N_EDGES 2097152
#define IN_DIM  256
#define HID     128
#define NCT     (HID / 16)    // 8 column tiles, fixed
#define N_SEG   320           // 32 * 10
#define MAXNORM (1.0f - 1e-5f)

// =====================================================================
// Pack W (K x N=128, f32 row-major) into bf16 WMMA B-fragments.
// Fragment layout per (kTile, colTile): 32 lanes x 16 halfs, where
// lane element j holds B[kTile*32 + (j<8 ? (lane>>4)*8 + j : 16 + (lane>>4)*8 + j-8)][colTile*16 + (lane&15)].
// =====================================================================
__global__ void pack_w(const float* __restrict__ W, unsigned short* __restrict__ frag,
                       int K, int N) {
    int idx = blockIdx.x * blockDim.x + threadIdx.x;
    int total = K * N;
    if (idx >= total) return;
    int j      = idx & 15;
    int lane   = (idx >> 4) & 31;
    int fragId = idx >> 9;                 // kTile * nct + colTile
    int nct    = N >> 4;
    int kt     = fragId / nct;
    int ct     = fragId - kt * nct;
    int k = kt * 32 + ((j < 8) ? ((lane >> 4) * 8 + j) : (16 + (lane >> 4) * 8 + (j - 8)));
    int n = ct * 16 + (lane & 15);
    frag[idx] = f2bf(W[(size_t)k * N + n]);
}

// =====================================================================
// GEMM: Out[M x 128] = X[M x K] (f32, cvt->bf16) @ Wfrag (bf16, swizzled)
// One wave computes one 16-row stripe across ALL 8 column tiles:
//   A fragment built once per K-step, then 8 back-to-back WMMAs.
// =====================================================================
__global__ __launch_bounds__(256) void gemm_bf16(
        const float* __restrict__ X, const unsigned short* __restrict__ Wfrag,
        float* __restrict__ Out, int M, int K) {
    int wave    = (int)((blockIdx.x * blockDim.x + threadIdx.x) >> 5);
    int lane    = threadIdx.x & 31;
    int rowTile = wave;
    if (rowTile * 16 >= M) return;          // wave-uniform -> EXEC all-ones inside

    int hi = lane >> 4;                     // 0 or 1
    int lo = lane & 15;
    int row = rowTile * 16 + lo;
    int kb  = hi * 8;
    const float* xr = X + (size_t)row * K;

    v8f acc[NCT];
    #pragma unroll
    for (int ct = 0; ct < NCT; ++ct)
        acc[ct] = (v8f){0.f, 0.f, 0.f, 0.f, 0.f, 0.f, 0.f, 0.f};

    int nkt = K >> 5;
    for (int kt = 0; kt < nkt; ++kt) {
        int kk = kt * 32;
        // A fragment (16-bit A 16x32 ISA layout), built ONCE per K-step
        float4 f0 = *reinterpret_cast<const float4*>(xr + kk + kb);
        float4 f1 = *reinterpret_cast<const float4*>(xr + kk + kb + 4);
        float4 f2 = *reinterpret_cast<const float4*>(xr + kk + kb + 16);
        float4 f3 = *reinterpret_cast<const float4*>(xr + kk + kb + 20);
        BF16x16 a;
        a.s[0]  = f2bf(f0.x); a.s[1]  = f2bf(f0.y); a.s[2]  = f2bf(f0.z); a.s[3]  = f2bf(f0.w);
        a.s[4]  = f2bf(f1.x); a.s[5]  = f2bf(f1.y); a.s[6]  = f2bf(f1.z); a.s[7]  = f2bf(f1.w);
        a.s[8]  = f2bf(f2.x); a.s[9]  = f2bf(f2.y); a.s[10] = f2bf(f2.z); a.s[11] = f2bf(f2.w);
        a.s[12] = f2bf(f3.x); a.s[13] = f2bf(f3.y); a.s[14] = f2bf(f3.z); a.s[15] = f2bf(f3.w);
        // 8 WMMAs reusing the A fragment (B fragments are L2-hot, 32B/lane)
        const unsigned short* bbase = Wfrag + ((size_t)kt * NCT << 9) + lane * 16;
        #pragma unroll
        for (int ct = 0; ct < NCT; ++ct) {
            BF16x16 b;
            const uint4* bp = reinterpret_cast<const uint4*>(bbase + (ct << 9));
            b.q[0] = bp[0];
            b.q[1] = bp[1];
            acc[ct] = __builtin_amdgcn_wmma_f32_16x16x32_bf16(
                          false, a.v, false, b.v, (short)0, acc[ct], false, false);
        }
    }
    // D layout: VGPR r -> M = r + 8*hi, N = lane&15 (per 16-col tile)
    float* orow = Out + (size_t)(rowTile * 16) * HID + lo;
    #pragma unroll
    for (int ct = 0; ct < NCT; ++ct) {
        #pragma unroll
        for (int r = 0; r < 8; ++r)
            orow[(size_t)(r + 8 * hi) * HID + ct * 16] = acc[ct][r];
    }
}

// =====================================================================
// Edge scatter: out[dst] += support[src]  (one wave per edge, 4 f32/lane)
// support is L2-resident (67 MB < 192 MB L2) -> gathers & atomics stay on-chip
// =====================================================================
__global__ __launch_bounds__(256) void scatter_edges(
        const float* __restrict__ sup, const int* __restrict__ src,
        const int* __restrict__ dst, float* __restrict__ out, int nEdges) {
    int e    = (int)((blockIdx.x * blockDim.x + threadIdx.x) >> 5);
    int lane = threadIdx.x & 31;
    if (e >= nEdges) return;
    int s = src[e];
    int d = dst[e];
    float4 v = *reinterpret_cast<const float4*>(sup + (size_t)s * HID + lane * 4);
    float* o = out + (size_t)d * HID + lane * 4;
    atomicAdd(o + 0, v.x);
    atomicAdd(o + 1, v.y);
    atomicAdd(o + 2, v.z);
    atomicAdd(o + 3, v.w);
}

// =====================================================================
// buf = tanh(buf + bias)   (elementwise, bias broadcast over HID=128)
// =====================================================================
__global__ void bias_tanh(float* __restrict__ buf, const float* __restrict__ b, int total) {
    int i = blockIdx.x * blockDim.x + threadIdx.x;
    if (i >= total) return;
    buf[i] = tanhf(buf[i] + b[i & (HID - 1)]);
}

// =====================================================================
// Per row: proj + logmap0 (fused scale), then segment-sum via atomics.
// One wave per node row (128 floats -> 4 per lane).
// =====================================================================
__global__ __launch_bounds__(256) void rownorm_segsum(
        const float* __restrict__ H, const int* __restrict__ seg,
        float* __restrict__ segsum, float* __restrict__ counts, int nNodes) {
    int row  = (int)((blockIdx.x * blockDim.x + threadIdx.x) >> 5);
    int lane = threadIdx.x & 31;
    if (row >= nNodes) return;
    float4 v = *reinterpret_cast<const float4*>(H + (size_t)row * HID + lane * 4);
    float ss = v.x * v.x + v.y * v.y + v.z * v.z + v.w * v.w;
    #pragma unroll
    for (int o = 16; o > 0; o >>= 1) ss += __shfl_xor(ss, o, 32);
    float norm = sqrtf(fmaxf(ss, 1e-15f));
    float s    = (norm > MAXNORM) ? (MAXNORM / norm) : 1.0f;   // proj
    float np   = norm * s;
    float arg  = fminf(np, 1.0f - 1e-7f);
    float fac  = s * atanhf(arg) / np;                          // logmap0 (c=1)
    int g = seg[row];
    float* o = segsum + (size_t)g * HID + lane * 4;
    atomicAdd(o + 0, v.x * fac);
    atomicAdd(o + 1, v.y * fac);
    atomicAdd(o + 2, v.z * fac);
    atomicAdd(o + 3, v.w * fac);
    if (lane == 0) atomicAdd(&counts[g], 1.0f);
}

// =====================================================================
// Finalize: agg = sum/max(cnt,1); proj(expmap0(agg)) -> d_out
// One wave per segment (320 segments).
// =====================================================================
__global__ __launch_bounds__(256) void finalize(
        const float* __restrict__ segsum, const float* __restrict__ counts,
        float* __restrict__ out, int nSeg) {
    int g    = (int)((blockIdx.x * blockDim.x + threadIdx.x) >> 5);
    int lane = threadIdx.x & 31;
    if (g >= nSeg) return;
    float inv = 1.0f / fmaxf(counts[g], 1.0f);
    float4 v = *reinterpret_cast<const float4*>(segsum + (size_t)g * HID + lane * 4);
    v.x *= inv; v.y *= inv; v.z *= inv; v.w *= inv;
    float ss = v.x * v.x + v.y * v.y + v.z * v.z + v.w * v.w;
    #pragma unroll
    for (int o = 16; o > 0; o >>= 1) ss += __shfl_xor(ss, o, 32);
    float norm = sqrtf(fmaxf(ss, 1e-15f));
    float fac  = tanhf(norm) / norm;                 // expmap0 (c=1)
    float rn   = norm * fac;                         // norm after expmap0
    float s2   = (rn > MAXNORM) ? (MAXNORM / rn) : 1.0f;   // proj
    fac *= s2;
    float4 r;
    r.x = v.x * fac; r.y = v.y * fac; r.z = v.z * fac; r.w = v.w * fac;
    *reinterpret_cast<float4*>(out + (size_t)g * HID + lane * 4) = r;
}

// =====================================================================
// Host launcher
// =====================================================================
extern "C" void kernel_launch(void* const* d_in, const int* in_sizes, int n_in,
                              void* d_out, int out_size, void* d_ws, size_t ws_size,
                              hipStream_t stream) {
    const float* x   = (const float*)d_in[0];
    const int*   src = (const int*)  d_in[1];
    const int*   dst = (const int*)  d_in[2];
    const int*   seg = (const int*)  d_in[3];
    const float* W1  = (const float*)d_in[4];
    const float* b1  = (const float*)d_in[5];
    const float* W2  = (const float*)d_in[6];
    const float* b2  = (const float*)d_in[7];
    float* out = (float*)d_out;

    char* ws = (char*)d_ws;
    const size_t SUP_BYTES = (size_t)N_NODES * HID * sizeof(float);   // 64 MB
    float*          bufA   = (float*)(ws);
    float*          bufB   = (float*)(ws + SUP_BYTES);
    unsigned short* wfrag1 = (unsigned short*)(ws + 2 * SUP_BYTES);
    unsigned short* wfrag2 = (unsigned short*)(ws + 2 * SUP_BYTES + IN_DIM * HID * 2);
    float*          segsum = (float*)(ws + 2 * SUP_BYTES + IN_DIM * HID * 2 + HID * HID * 2);
    float*          counts = (float*)((char*)segsum + (size_t)N_SEG * HID * sizeof(float));

    // --- pack weights into bf16 WMMA B-fragments ---
    pack_w<<<(IN_DIM * HID + 255) / 256, 256, 0, stream>>>(W1, wfrag1, IN_DIM, HID);
    pack_w<<<(HID * HID + 255) / 256, 256, 0, stream>>>(W2, wfrag2, HID, HID);

    const int gemmBlocks = (N_NODES / 16) * 32 / 256;   // one wave per 16-row stripe

    // --- layer 1: support = x @ W1 ---
    gemm_bf16<<<gemmBlocks, 256, 0, stream>>>(x, wfrag1, bufA, N_NODES, IN_DIM);
    hipMemsetAsync(bufB, 0, SUP_BYTES, stream);
    scatter_edges<<<(N_EDGES * 32 + 255) / 256, 256, 0, stream>>>(bufA, src, dst, bufB, N_EDGES);
    bias_tanh<<<(N_NODES * HID + 255) / 256, 256, 0, stream>>>(bufB, b1, N_NODES * HID);

    // --- layer 2: support = h @ W2 ---
    gemm_bf16<<<gemmBlocks, 256, 0, stream>>>(bufB, wfrag2, bufA, N_NODES, HID);
    hipMemsetAsync(bufB, 0, SUP_BYTES, stream);
    scatter_edges<<<(N_EDGES * 32 + 255) / 256, 256, 0, stream>>>(bufA, src, dst, bufB, N_EDGES);
    bias_tanh<<<(N_NODES * HID + 255) / 256, 256, 0, stream>>>(bufB, b2, N_NODES * HID);

    // --- hyperbolic tangent-space pooling ---
    hipMemsetAsync(segsum, 0, (size_t)N_SEG * HID * sizeof(float) + N_SEG * sizeof(float), stream);
    rownorm_segsum<<<(N_NODES * 32 + 255) / 256, 256, 0, stream>>>(bufB, seg, segsum, counts, N_NODES);
    finalize<<<(N_SEG * 32 + 255) / 256, 256, 0, stream>>>(segsum, counts, out, N_SEG);
}
